// AttLayer_67353677136176
// MI455X (gfx1250) — matrix operand order repaired and checked by reading, not compiled, not run on hardware
//
#include <hip/hip_runtime.h>

// Flash-attention formulation of: xt = X X^T (symmetric) ; ait = softmax(xt, axis=1)
// ; out[b,j,d] = sum_i x[b,i,d]*ait[b,i,j]  ==  out = softmax_rows(X X^T) X.
// B=16, S=2048, D=512, fp32 in/out; bf16 WMMA operands, fp32 accum.

#define BB   16
#define SS   2048
#define DD   512
#define MQ   64          // queries per workgroup
#define NK   32          // keys per inner block
#define NTHREADS 256     // 8 waves (wave32)
#define QSTR 520         // LDS row stride (halves) for Q (512 + 8 pad)
#define KSTR 520         // LDS row stride (halves) for K block
#define VTSTR 40         // LDS row stride (halves) for V^T (32 + 8 pad, 16B aligned)
#define PSTR  40         // LDS row stride (halves) for P (32 + 8 pad, 16B aligned)

typedef __attribute__((ext_vector_type(8)))  __bf16 v8bf;
typedef __attribute__((ext_vector_type(16))) __bf16 v16bf;
typedef __attribute__((ext_vector_type(8)))  float  v8f;

__device__ __forceinline__ unsigned short f2bf(float f) {
    unsigned u = __float_as_uint(f);
    return (unsigned short)((u + 0x7FFFu + ((u >> 16) & 1u)) >> 16); // RNE truncate to bf16
}

__device__ __forceinline__ v8f zero8() {
    v8f r;
#pragma unroll
    for (int i = 0; i < 8; ++i) r[i] = 0.0f;
    return r;
}

__device__ __forceinline__ v16bf cat16(v8bf lo, v8bf hi) {
    v16bf r;
#pragma unroll
    for (int i = 0; i < 8; ++i) { r[i] = lo[i]; r[i + 8] = hi[i]; }
    return r;
}

__device__ __forceinline__ v16bf ld16(const unsigned short* p) {
    // two 16B ds_load_b128's
    v8bf lo = *(const v8bf*)p;
    v8bf hi = *(const v8bf*)(p + 8);
    return cat16(lo, hi);
}

__global__ __launch_bounds__(NTHREADS, 1)
void attn_flash_kernel(const float* __restrict__ x, float* __restrict__ out) {
    extern __shared__ unsigned short smem[];
    unsigned short* qs = smem;                    // MQ x QSTR  (Q rows, bf16)
    unsigned short* ks = qs + MQ * QSTR;          // NK x KSTR  (K rows, bf16)
    unsigned short* vt = ks + NK * KSTR;          // DD x VTSTR (V transposed, bf16)
    unsigned short* pl = vt + DD * VTSTR;         // MQ x PSTR  (P rows, bf16)
    float* fst  = (float*)(pl + MQ * PSTR);
    float* m_s  = fst;          // [64] running row max
    float* l_s  = fst + 64;     // [64] running row sum
    float* sc_s = fst + 128;    // [64] per-block rescale factor (and final 1/l)
    float* pmax = fst + 192;    // [2][64] per-ktile partial max
    float* psum = fst + 320;    // [2][64] per-ktile partial sums

    const int tid  = threadIdx.x;
    const int lane = tid & 31;
    const int w    = tid >> 5;          // wave id 0..7
    const int b    = blockIdx.y;
    const int q0   = blockIdx.x * MQ;
    const int hi   = lane >> 4;         // lane half: 0/1
    const int ln   = lane & 15;

    if (tid < MQ) { m_s[tid] = -1e30f; l_s[tid] = 0.0f; }

    // ---- load Q block (MQ x DD fp32 -> bf16 in LDS), coalesced float2 ----
    {
        const float* xq = x + ((size_t)b * SS + q0) * DD;
        for (int i = tid; i < MQ * (DD / 2); i += NTHREADS) {
            int row = i >> 8;           // DD/2 = 256 pairs per row
            int dp  = i & 255;
            float2 f = *(const float2*)(xq + (size_t)row * DD + dp * 2);
            *(unsigned*)(qs + row * QSTR + dp * 2) =
                (unsigned)f2bf(f.x) | ((unsigned)f2bf(f.y) << 16);
        }
    }
    __syncthreads();

    // O accumulators: 4 q-tiles x 4 d-tiles; this wave owns d in [w*64, w*64+64)
    v8f O[4][4];
#pragma unroll
    for (int qt = 0; qt < 4; ++qt)
#pragma unroll
        for (int dt = 0; dt < 4; ++dt) O[qt][dt] = zero8();

    const int ktile = w >> 2;   // 0..1 : which 16-key slice of S^T this wave computes
    const int qtile = w & 3;    // 0..3 : which 16-query slice

    for (int k0 = 0; k0 < SS; k0 += NK) {
        // ---- Phase A: K/V block (NK x DD) -> ks (row major) + vt (transposed) ----
        const float* xk = x + ((size_t)b * SS + k0) * DD;
        for (int i = tid; i < NK * (DD / 2); i += NTHREADS) {
            int row = i >> 8;
            int dp  = i & 255;
            float2 f = *(const float2*)(xk + (size_t)row * DD + dp * 2);
            unsigned short b0 = f2bf(f.x), b1 = f2bf(f.y);
            *(unsigned*)(ks + row * KSTR + dp * 2) = (unsigned)b0 | ((unsigned)b1 << 16);
            vt[(dp * 2    ) * VTSTR + row] = b0;
            vt[(dp * 2 + 1) * VTSTR + row] = b1;
        }
        __syncthreads();

        // ---- Phase B: S^T tile (16k x 16q) = K[ktile] * Q^T[qtile], fp32 acc ----
        v8f acc = zero8();
        const unsigned short* arow = ks + (ktile * 16 + ln) * KSTR + hi * 8;   // A: rows = keys
        const unsigned short* brow = qs + (qtile * 16 + ln) * QSTR + hi * 8;   // B: cols = queries
#pragma unroll
        for (int dc = 0; dc < DD / 32; ++dc) {
            v16bf a  = cat16(*(const v8bf*)(arow + dc * 32),
                             *(const v8bf*)(arow + dc * 32 + 16));
            v16bf bb = cat16(*(const v8bf*)(brow + dc * 32),
                             *(const v8bf*)(brow + dc * 32 + 16));
            acc = __builtin_amdgcn_wmma_f32_16x16x32_bf16(false, a, false, bb,
                                                          (short)0, acc, false, false);
        }
        // per-lane column (=query) max over this tile's 16 keys
        float mx = acc[0];
#pragma unroll
        for (int r = 1; r < 8; ++r) mx = fmaxf(mx, acc[r]);
        mx = fmaxf(mx, __shfl_xor(mx, 16, 32));
        if (lane < 16) pmax[ktile * 64 + qtile * 16 + lane] = mx;
        __syncthreads();

        // ---- Phase C: wave0 combines partial maxes, computes rescale factors ----
        if (w == 0) {
            for (int q = lane; q < MQ; q += 32) {
                float bm = fmaxf(pmax[q], pmax[64 + q]);
                float mo = m_s[q];
                float mn = fmaxf(mo, bm);
                m_s[q]  = mn;
                sc_s[q] = __expf(mo - mn);
            }
        }
        __syncthreads();

        // ---- Phase D: P = exp(S^T - m), bf16 -> pl; partial sums; rescale O ----
        {
            float mq = m_s[qtile * 16 + ln];     // whole lane shares one query column
            float p[8]; float ps = 0.0f;
#pragma unroll
            for (int r = 0; r < 8; ++r) { p[r] = __expf(acc[r] - mq); ps += p[r]; }
            uint4 pk;
            pk.x = (unsigned)f2bf(p[0]) | ((unsigned)f2bf(p[1]) << 16);
            pk.y = (unsigned)f2bf(p[2]) | ((unsigned)f2bf(p[3]) << 16);
            pk.z = (unsigned)f2bf(p[4]) | ((unsigned)f2bf(p[5]) << 16);
            pk.w = (unsigned)f2bf(p[6]) | ((unsigned)f2bf(p[7]) << 16);
            // rows k = ktile*16 + hi*8 + r, stored row-major by query -> WMMA-A layout
            *(uint4*)(pl + (qtile * 16 + ln) * PSTR + ktile * 16 + hi * 8) = pk;
            ps += __shfl_xor(ps, 16, 32);
            if (lane < 16) psum[ktile * 64 + qtile * 16 + lane] = ps;

            // rescale O accumulators by exp(m_old - m_new) per query row
#pragma unroll
            for (int qt = 0; qt < 4; ++qt) {
                float4 s0 = *(const float4*)&sc_s[qt * 16 + hi * 8];
                float4 s1 = *(const float4*)&sc_s[qt * 16 + hi * 8 + 4];
                float sr[8] = {s0.x, s0.y, s0.z, s0.w, s1.x, s1.y, s1.z, s1.w};
#pragma unroll
                for (int dt = 0; dt < 4; ++dt)
#pragma unroll
                    for (int r = 0; r < 8; ++r) O[qt][dt][r] *= sr[r];
            }
        }
        __syncthreads();

        // ---- Phase E: wave0 updates l; all waves accumulate O += P * V ----
        if (w == 0) {
            for (int q = lane; q < MQ; q += 32)
                l_s[q] = l_s[q] * sc_s[q] + psum[q] + psum[64 + q];
        }
        {
            v16bf Bop[4];
#pragma unroll
            for (int dt = 0; dt < 4; ++dt)   // B cols = d, K-dim = keys, contiguous in vt
                Bop[dt] = cat16(*(const v8bf*)(vt + (w * 64 + dt * 16 + ln) * VTSTR + hi * 8),
                                *(const v8bf*)(vt + (w * 64 + dt * 16 + ln) * VTSTR + hi * 8 + 16));
#pragma unroll
            for (int qt = 0; qt < 4; ++qt) {
                const unsigned short* pr = pl + (qt * 16 + ln) * PSTR + hi * 8;
                v16bf Aop = cat16(*(const v8bf*)pr, *(const v8bf*)(pr + 16));
#pragma unroll
                for (int dt = 0; dt < 4; ++dt)
                    O[qt][dt] = __builtin_amdgcn_wmma_f32_16x16x32_bf16(
                        false, Aop, false, Bop[dt], (short)0, O[qt][dt], false, false);
            }
        }
        __syncthreads();
    }

    // ---- finalize: divide by l, write out ----
    if (w == 0) {
        for (int q = lane; q < MQ; q += 32) sc_s[q] = 1.0f / l_s[q];
    }
    __syncthreads();

    float* op = out + ((size_t)b * SS + q0) * DD;
#pragma unroll
    for (int qt = 0; qt < 4; ++qt) {
        float4 s0 = *(const float4*)&sc_s[qt * 16 + hi * 8];
        float4 s1 = *(const float4*)&sc_s[qt * 16 + hi * 8 + 4];
        float sr[8] = {s0.x, s0.y, s0.z, s0.w, s1.x, s1.y, s1.z, s1.w};
#pragma unroll
        for (int dt = 0; dt < 4; ++dt) {
            int dcol = w * 64 + dt * 16 + ln;
#pragma unroll
            for (int r = 0; r < 8; ++r) {
                int qrow = qt * 16 + hi * 8 + r;
                op[(size_t)qrow * DD + dcol] = O[qt][dt][r] * sr[r];
            }
        }
    }
}

extern "C" void kernel_launch(void* const* d_in, const int* in_sizes, int n_in,
                              void* d_out, int out_size, void* d_ws, size_t ws_size,
                              hipStream_t stream) {
    (void)in_sizes; (void)n_in; (void)out_size; (void)d_ws; (void)ws_size;
    const float* x = (const float*)d_in[0];
    float* out = (float*)d_out;

    const size_t shmem = (size_t)(MQ * QSTR + NK * KSTR + DD * VTSTR + MQ * PSTR) * 2
                       + 448 * sizeof(float);   // 147,712 bytes, < 320KB/WGP
    hipFuncSetAttribute((const void*)attn_flash_kernel,
                        hipFuncAttributeMaxDynamicSharedMemorySize, (int)shmem);

    dim3 grid(SS / MQ, BB);     // 32 x 16 = 512 workgroups
    attn_flash_kernel<<<grid, NTHREADS, shmem, stream>>>(x, out);
}
